// AttentionLayer_20985210208983
// MI455X (gfx1250) — matrix operand order repaired and checked by reading, not compiled, not run on hardware
//
#include <hip/hip_runtime.h>

// ---------------------------------------------------------------------------
// MI455X (gfx1250, wave32) attention:   out = softmax(Q^T K) V  (1x1-conv QKV)
//   x : (B, Cin, N) f32     Wq,Wk : (Ck, Cin) f32    Wv : (Co, Cin) f32
//   B=8, Cin=512, N=2048, Ck=256, Co=512
// bf16 WMMA everywhere; flash-attention single pass with 64-row query tiles
// (keeps K/V L2 re-read traffic ~0.75 GB, balanced vs ~14us of WMMA compute).
// ---------------------------------------------------------------------------

#define B_SZ   8
#define CIN    512
#define NPOS   2048
#define CK     256
#define CO     512
#define NT     64      // query rows per block (attention)
#define MT     32      // key positions per m-iteration

typedef __attribute__((ext_vector_type(16))) __bf16 v16bf;
typedef __attribute__((ext_vector_type(8)))  float  v8f;

union Frag16 {
    v16bf          v;
    unsigned short u[16];
    uint4          q[2];   // two 16-byte chunks -> global/ds *_b128
};

__device__ __forceinline__ unsigned short f2bf(float f) {
    unsigned int u = __builtin_bit_cast(unsigned int, f);
    u += 0x7FFFu + ((u >> 16) & 1u);          // round-to-nearest-even
    return (unsigned short)(u >> 16);
}

// ---------------------------------------------------------------------------
// Kernel 1: Y[n, c'] = sum_c W[c', c] * x[b, c, n]   (1x1 conv == GEMM)
// Block tile 64(n) x 64(c'), K-step 32. 8 waves = 4 n-subtiles x 2 c'-subtiles.
// MODE 0: Y as (B, N, Cout) bf16   (Q^T/K^T layout, k contiguous)
// MODE 1: Y as (B, Cout, N) bf16   (V layout, n contiguous)
// ---------------------------------------------------------------------------
template<int MODE>
__global__ __launch_bounds__(256)
void qkv_proj_kernel(const float* __restrict__ x, const float* __restrict__ W,
                     unsigned short* __restrict__ Y, int Cout)
{
    __shared__ unsigned short xs[32][64];          // x tile [c][n], bf16

    const int b    = blockIdx.z;
    const int n0   = blockIdx.x * 64;
    const int c0o  = blockIdx.y * 64;
    const int tid  = threadIdx.x;
    const int lane = tid & 31;
    const int wave = tid >> 5;
    const int wn   = wave & 3;                     // n-subtile (16 rows)
    const int wc   = wave >> 2;                    // c'-subtile (32 cols)
    const int l16  = lane & 15;
    const int lhi  = lane >> 4;

    const float* xb = x + (size_t)b * CIN * NPOS;

    v8f acc0 = {};
    v8f acc1 = {};

    for (int k0 = 0; k0 < CIN; k0 += 32) {
        // ---- stage x[b, k0..k0+31, n0..n0+63] -> LDS bf16 (coalesced) ----
        {
            const int row = tid >> 3;              // 0..31  (c within chunk)
            const int col = (tid & 7) * 8;         // 0..56  (n within tile)
            const float* src = xb + (size_t)(k0 + row) * NPOS + n0 + col;
            float4 a0 = *(const float4*)(src);
            float4 a1 = *(const float4*)(src + 4);
            xs[row][col + 0] = f2bf(a0.x);  xs[row][col + 1] = f2bf(a0.y);
            xs[row][col + 2] = f2bf(a0.z);  xs[row][col + 3] = f2bf(a0.w);
            xs[row][col + 4] = f2bf(a1.x);  xs[row][col + 5] = f2bf(a1.y);
            xs[row][col + 6] = f2bf(a1.z);  xs[row][col + 7] = f2bf(a1.w);
        }
        __syncthreads();

        // ---- A fragment: rows = n, K = c (contiguous k = 16*lhi + e) ----
        Frag16 A;
        const int nloc = wn * 16 + l16;
        #pragma unroll
        for (int e = 0; e < 16; ++e) {
            const int kk = (lhi << 4) + e;
            A.u[e] = xs[kk][nloc];
        }

        // ---- B fragments straight from W (W columns contiguous in c) ----
        Frag16 B0, B1;
        {
            const float* wp0 = W + (size_t)(c0o + wc * 32 + l16) * CIN + k0 + lhi * 16;
            const float* wp1 = wp0 + (size_t)16 * CIN;
            #pragma unroll
            for (int e = 0; e < 16; ++e) {
                B0.u[e] = f2bf(wp0[e]);
                B1.u[e] = f2bf(wp1[e]);
            }
        }

        acc0 = __builtin_amdgcn_wmma_f32_16x16x32_bf16(false, A.v, false, B0.v,
                                                       (short)0, acc0, false, false);
        acc1 = __builtin_amdgcn_wmma_f32_16x16x32_bf16(false, A.v, false, B1.v,
                                                       (short)0, acc1, false, false);
        __syncthreads();
    }

    // ---- store D tiles (row = n0+wn*16+r+8*lhi, col = c0o+wc*32+l16[+16]) ----
    #pragma unroll
    for (int r = 0; r < 8; ++r) {
        const int n  = n0 + wn * 16 + r + lhi * 8;
        const int c0 = c0o + wc * 32 + l16;
        if (MODE == 0) {
            unsigned short* yp = Y + ((size_t)b * NPOS + n) * (size_t)Cout;
            yp[c0]      = f2bf(acc0[r]);
            yp[c0 + 16] = f2bf(acc1[r]);
        } else {
            unsigned short* yp = Y + (size_t)b * Cout * NPOS + n;
            yp[(size_t)c0 * NPOS]        = f2bf(acc0[r]);
            yp[(size_t)(c0 + 16) * NPOS] = f2bf(acc1[r]);
        }
    }
}

// ---------------------------------------------------------------------------
// Kernel 2: flash attention, 64 query rows per block, 8 waves.
// Per m-tile (32 keys): wave w (ns=w&3, ms=w>>2) computes one 16x16 S frag
// (k=256, 8 WMMAs, Q from LDS), online softmax across waves, P -> LDS bf16,
// then each wave does 16 WMMAs of P x V^T into its 64n x 64o accumulator.
// ---------------------------------------------------------------------------
__global__ __launch_bounds__(256)
void attn_kernel(const unsigned short* __restrict__ Qt,  // (B, N, CK) bf16
                 const unsigned short* __restrict__ Kt,  // (B, N, CK) bf16
                 const unsigned short* __restrict__ Vb,  // (B, CO, N) bf16
                 float* __restrict__ out)                // (B, CO, N) f32
{
    __shared__ unsigned short Qs[NT][CK + 8];   // stride 528 B (16B-aligned)
    __shared__ unsigned short Pl[NT][MT + 8];   // stride 80 B  (16B-aligned)
    __shared__ float pmax[2][NT], psum[2][NT];
    __shared__ float row_max[NT], row_sum[NT], row_scale[NT];

    const int b    = blockIdx.y;
    const int n0   = blockIdx.x * NT;
    const int tid  = threadIdx.x;
    const int lane = tid & 31;
    const int wave = tid >> 5;
    const int l16  = lane & 15;
    const int lhi  = lane >> 4;
    const int ns   = wave & 3;                  // S-stage n-subtile (16 rows)
    const int ms   = wave >> 2;                 // S-stage m-subtile (16 keys)

    const unsigned short* Qb = Qt + (size_t)b * NPOS * CK;
    const unsigned short* Kb = Kt + (size_t)b * NPOS * CK;
    const unsigned short* Vp = Vb + (size_t)b * CO * NPOS;

    // ---- stage Q tile (64 x 256 bf16) into LDS, coalesced b128 copies ----
    {
        const int r  = tid >> 2;                // 0..63
        const int c0 = (tid & 3) * 64;          // half-offset within row
        const uint4* src = (const uint4*)(Qb + (size_t)(n0 + r) * CK + c0);
        uint4* dst = (uint4*)&Qs[r][c0];
        #pragma unroll
        for (int j = 0; j < 8; ++j) dst[j] = src[j];
    }
    if (tid < NT) { row_max[tid] = -3.0e38f; row_sum[tid] = 0.0f; }
    __syncthreads();

    const v8f vzero = {};
    v8f Oacc[4][4];                             // [n-subtile][o-subtile]
    #pragma unroll
    for (int i = 0; i < 4; ++i)
        #pragma unroll
        for (int j = 0; j < 4; ++j) Oacc[i][j] = vzero;

    for (int i = 0; i < NPOS / MT; ++i) {
        const int m0 = i * MT;

        // ---- prefetch next iteration's K/V rows (global_prefetch_b8) ----
        if (i + 1 < NPOS / MT) {
            __builtin_prefetch(Kb + (size_t)(m0 + MT + ms * 16 + l16) * CK, 0, 2);
            __builtin_prefetch(Vp + (size_t)(wave * 64 + lane) * NPOS + m0 + MT, 0, 2);
            __builtin_prefetch(Vp + (size_t)(wave * 64 + 32 + lane) * NPOS + m0 + MT, 0, 2);
        }

        // ---- S = Q^T K : one 16x16 fragment per wave, k = 256 ----
        v8f S = {};
        const unsigned short* krow = Kb + (size_t)(m0 + ms * 16 + l16) * CK;
        #pragma unroll
        for (int kc = 0; kc < 8; ++kc) {
            Frag16 Aq, Bk;
            const unsigned short* qsrow = &Qs[ns * 16 + l16][kc * 32 + lhi * 8];
            Aq.q[0] = *(const uint4*)(qsrow);
            Aq.q[1] = *(const uint4*)(qsrow + 16);
            Bk.q[0] = *(const uint4*)(krow + kc * 32 + lhi * 8);
            Bk.q[1] = *(const uint4*)(krow + kc * 32 + lhi * 8 + 16);
            S = __builtin_amdgcn_wmma_f32_16x16x32_bf16(false, Aq.v, false, Bk.v,
                                                        (short)0, S, false, false);
        }

        // ---- local row max (16-lane groups each hold one row) ----
        #pragma unroll
        for (int r = 0; r < 8; ++r) {
            float v = S[r];
            #pragma unroll
            for (int off = 1; off < 16; off <<= 1)
                v = fmaxf(v, __shfl_xor(v, off, 32));
            if (l16 == 0) pmax[ms][ns * 16 + r + lhi * 8] = v;
        }
        __syncthreads();

        if (tid < NT) {                         // combine maxes, prep rescale
            const float m_old = row_max[tid];
            const float mx = fmaxf(m_old, fmaxf(pmax[0][tid], pmax[1][tid]));
            row_max[tid]   = mx;
            row_scale[tid] = __expf(m_old - mx);
        }
        __syncthreads();

        // ---- P = exp(S - max) -> LDS ; partial row sums ; rescale O ----
        #pragma unroll
        for (int r = 0; r < 8; ++r) {
            const int row = ns * 16 + r + lhi * 8;
            const float p = __expf(S[r] - row_max[row]);
            Pl[row][ms * 16 + l16] = f2bf(p);
            float s = p;
            #pragma unroll
            for (int off = 1; off < 16; off <<= 1)
                s += __shfl_xor(s, off, 32);
            if (l16 == 0) psum[ms][row] = s;
        }
        #pragma unroll
        for (int nsub = 0; nsub < 4; ++nsub)
            #pragma unroll
            for (int r = 0; r < 8; ++r) {
                const float sc = row_scale[nsub * 16 + r + lhi * 8];
                #pragma unroll
                for (int osub = 0; osub < 4; ++osub)
                    Oacc[nsub][osub][r] *= sc;
            }
        __syncthreads();

        if (tid < NT)
            row_sum[tid] = row_sum[tid] * row_scale[tid] + psum[0][tid] + psum[1][tid];

        // ---- O += P x V^T : A frags from LDS, B frags contiguous in V ----
        #pragma unroll
        for (int nsub = 0; nsub < 4; ++nsub) {
            Frag16 Ap;
            const unsigned short* prow = &Pl[nsub * 16 + l16][lhi * 8];
            Ap.q[0] = *(const uint4*)(prow);
            Ap.q[1] = *(const uint4*)(prow + 16);
            #pragma unroll
            for (int osub = 0; osub < 4; ++osub) {
                Frag16 Bv;
                const unsigned short* vrow =
                    Vp + (size_t)(wave * 64 + osub * 16 + l16) * NPOS + m0 + lhi * 8;
                Bv.q[0] = *(const uint4*)(vrow);
                Bv.q[1] = *(const uint4*)(vrow + 16);
                Oacc[nsub][osub] = __builtin_amdgcn_wmma_f32_16x16x32_bf16(
                    false, Ap.v, false, Bv.v, (short)0, Oacc[nsub][osub], false, false);
            }
        }
        __syncthreads();                        // protect Pl before next overwrite
    }

    // ---- normalize (multiply by 1/row_sum) and store as float4 runs ----
    if (tid < NT) row_scale[tid] = 1.0f / row_sum[tid];
    __syncthreads();

    #pragma unroll
    for (int nsub = 0; nsub < 4; ++nsub)
        #pragma unroll
        for (int osub = 0; osub < 4; ++osub) {
            const int oc    = wave * 64 + osub * 16 + l16;
            const int nbase = n0 + nsub * 16 + lhi * 8;     // rows r=0..7 consecutive
            float* op = out + ((size_t)b * CO + oc) * NPOS + nbase;
            float4 v0, v1;
            v0.x = Oacc[nsub][osub][0] * row_scale[nsub * 16 + lhi * 8 + 0];
            v0.y = Oacc[nsub][osub][1] * row_scale[nsub * 16 + lhi * 8 + 1];
            v0.z = Oacc[nsub][osub][2] * row_scale[nsub * 16 + lhi * 8 + 2];
            v0.w = Oacc[nsub][osub][3] * row_scale[nsub * 16 + lhi * 8 + 3];
            v1.x = Oacc[nsub][osub][4] * row_scale[nsub * 16 + lhi * 8 + 4];
            v1.y = Oacc[nsub][osub][5] * row_scale[nsub * 16 + lhi * 8 + 5];
            v1.z = Oacc[nsub][osub][6] * row_scale[nsub * 16 + lhi * 8 + 6];
            v1.w = Oacc[nsub][osub][7] * row_scale[nsub * 16 + lhi * 8 + 7];
            *(float4*)(op)     = v0;
            *(float4*)(op + 4) = v1;
        }
}

// ---------------------------------------------------------------------------
extern "C" void kernel_launch(void* const* d_in, const int* in_sizes, int n_in,
                              void* d_out, int out_size, void* d_ws, size_t ws_size,
                              hipStream_t stream)
{
    (void)in_sizes; (void)n_in; (void)out_size; (void)ws_size;

    const float* x  = (const float*)d_in[0];
    const float* Wq = (const float*)d_in[1];
    const float* Wk = (const float*)d_in[2];
    const float* Wv = (const float*)d_in[3];
    float* out = (float*)d_out;

    // workspace: Qt (8MB) | Kt (8MB) | V (16MB), all bf16 (ushort)
    unsigned short* Qt = (unsigned short*)d_ws;
    unsigned short* Kt = Qt + (size_t)B_SZ * NPOS * CK;
    unsigned short* Vw = Kt + (size_t)B_SZ * NPOS * CK;

    dim3 blk(256);
    qkv_proj_kernel<0><<<dim3(NPOS / 64, CK / 64, B_SZ), blk, 0, stream>>>(x, Wq, Qt, CK);
    qkv_proj_kernel<0><<<dim3(NPOS / 64, CK / 64, B_SZ), blk, 0, stream>>>(x, Wk, Kt, CK);
    qkv_proj_kernel<1><<<dim3(NPOS / 64, CO / 64, B_SZ), blk, 0, stream>>>(x, Wv, Vw, CO);
    attn_kernel<<<dim3(NPOS / NT, B_SZ), blk, 0, stream>>>(Qt, Kt, Vw, out);
}
